// AttentionBlock_644245095146
// MI455X (gfx1250) — matrix operand order repaired and checked by reading, not compiled, not run on hardware
//
#include <hip/hip_runtime.h>

// ---------------------------------------------------------------------------
// AttentionBlock for MI455X (gfx1250, wave32, WMMA 16x16x32 f16 -> f32)
// B=8, C=512, N=4096, 32 groups.
//  - All GEMMs through v_wmma_f32_16x16x32_f16.
//  - Panel-shaped operands (weights, K rows, P chunks) are staged to LDS by
//    the Tensor Data Mover, double-buffered against s_wait_tensorcnt so the
//    DMA of panel i+1 overlaps WMMA on panel i; WMMA reads them via ds_load.
//  - pv uses 64-row blocks with 8 independent B-load chains per k-step so a
//    single s_wait covers 8 WMMAs.
// ---------------------------------------------------------------------------

#define B_DIM 8
#define C_DIM 512
#define N_DIM 4096
#define LDSP  520            // padded LDS row stride (halfs): 512 + 8
#define QSCALE 0.04419417382415922f   // 512^-0.5

typedef _Float16 half16 __attribute__((ext_vector_type(16)));
typedef _Float16 half8  __attribute__((ext_vector_type(8)));
typedef float    float8 __attribute__((ext_vector_type(8)));
typedef unsigned int u32x4 __attribute__((ext_vector_type(4)));
typedef int i32x4 __attribute__((ext_vector_type(4)));
typedef int i32x8 __attribute__((ext_vector_type(8)));

#if defined(__has_builtin)
#if __has_builtin(__builtin_amdgcn_tensor_load_to_lds) && __has_builtin(__builtin_amdgcn_s_wait_tensorcnt)
#define HAVE_TDM 1
#endif
#endif
#ifndef HAVE_TDM
#define HAVE_TDM 0
#endif

// ---- WMMA operand loaders --------------------------------------------------
// A (16x32 f16): lanes 0-15 -> row=lane, K {0..15};
//                lanes 16-31 -> row=lane-16, K {16..31}.
__device__ __forceinline__ half16 load_A_lds(const _Float16* As, int k0, int lane) {
    int r    = lane & 15;
    int koff = k0 + (lane >> 4) * 16;
    const _Float16* p = As + r * LDSP + koff;
    half8 lo = *(const half8*)(p);
    half8 hi = *(const half8*)(p + 8);
    return __builtin_shufflevector(lo, hi, 0,1,2,3,4,5,6,7,8,9,10,11,12,13,14,15);
}

// B (32x16 f16): lane -> col = lane&15, 16 contiguous K starting at (lane>>4)*16.
__device__ __forceinline__ half16 load_B_glb(const _Float16* base, int stride, int lane) {
    int col = lane & 15;
    int kb  = (lane >> 4) * 16;
    const _Float16* p = base + (size_t)col * stride + kb;
    half8 lo = *(const half8*)(p);
    half8 hi = *(const half8*)(p + 8);
    return __builtin_shufflevector(lo, hi, 0,1,2,3,4,5,6,7,8,9,10,11,12,13,14,15);
}

__device__ __forceinline__ half16 load_B_lds(const _Float16* base, int k0, int lane) {
    int col = lane & 15;
    int kb  = k0 + (lane >> 4) * 16;
    const _Float16* p = base + col * LDSP + kb;
    half8 lo = *(const half8*)(p);
    half8 hi = *(const half8*)(p + 8);
    return __builtin_shufflevector(lo, hi, 0,1,2,3,4,5,6,7,8,9,10,11,12,13,14,15);
}

__device__ __forceinline__ float8 wmma_f16(half16 a, half16 b, float8 c) {
    return __builtin_amdgcn_wmma_f32_16x16x32_f16(false, a, false, b, (short)0, c, false, false);
}

// ---- Tile staging ----------------------------------------------------------
// Stage rows x 512-half tile (global row stride = gstride halfs) into LDS with
// row stride LDSP (TDM pads 4 dwords after every 256 dwords).
__device__ __forceinline__ void tdm_stage_rows(void* lds_dst, const _Float16* gsrc,
                                               unsigned rows, unsigned gstride) {
#if HAVE_TDM
    unsigned long long ga = (unsigned long long)(__UINTPTR_TYPE__)gsrc;
    unsigned lds = (unsigned)(__UINTPTR_TYPE__)lds_dst;   // generic low 32 = LDS offset
    u32x4 g0;
    g0.x = 1u;                                            // count=1, user descriptor
    g0.y = lds;                                           // lds_addr
    g0.z = (unsigned)ga;                                  // global_addr[31:0]
    g0.w = (unsigned)((ga >> 32) & 0x01FFFFFFu) | (2u << 30);  // addr[56:32] | type=2
    unsigned td0 = gstride;
    unsigned td1 = 1u << 20;
    i32x8 g1;
    g1[0] = (int)((1u << 16)            // data_size = 2 bytes
                | (1u << 20)            // pad_enable
                | (7u << 22)            // pad_interval: 256 dwords
                | (3u << 25));          // pad_amount: 4 dwords (8 halfs)
    g1[1] = (int)((td0 & 0xffffu) << 16);
    g1[2] = (int)(((td0 >> 16) & 0xffffu) | ((td1 & 0xffffu) << 16));
    g1[3] = (int)(((td1 >> 16) & 0xffffu) | (512u << 16));          // tile_dim0 = 512
    g1[4] = (int)(rows & 0xffffu);                                  // tile_dim1
    g1[5] = (int)td0;                                               // dim0_stride lo32
    g1[6] = 0;
    g1[7] = 0;
    i32x4 z4 = {0, 0, 0, 0};
#if __clang_major__ >= 23
    i32x8 z8 = {0, 0, 0, 0, 0, 0, 0, 0};
    __builtin_amdgcn_tensor_load_to_lds(g0, g1, z4, z4, z8, 0);
#else
    __builtin_amdgcn_tensor_load_to_lds(g0, g1, z4, z4, 0);
#endif
#endif
}

__device__ __forceinline__ void manual_stage_rows(_Float16* dst, const _Float16* gsrc,
                                                  int rows, int gstride, int t, int nthr) {
    int total = rows * 16;
    for (int idx = t; idx < total; idx += nthr) {
        int r = idx >> 4, c = (idx & 15) * 32;
        const _Float16* s = gsrc + (size_t)r * gstride + c;
        *(half16*)&dst[r * LDSP + c]      = *(const half16*)(s);
        *(half16*)&dst[r * LDSP + c + 16] = *(const half16*)(s + 16);
    }
}

// ---------------------------------------------------------------------------
// 1) GroupNorm -> hT [B, N, C] f16 (transposed)
// ---------------------------------------------------------------------------
__global__ __launch_bounds__(256) void gn_kernel(const float* __restrict__ x,
                                                 const float* __restrict__ gamma,
                                                 const float* __restrict__ beta,
                                                 _Float16* __restrict__ hT) {
    const int cpg = C_DIM / 32;
    int b = blockIdx.x >> 5, g = blockIdx.x & 31;
    const float* xg = x + ((size_t)b * C_DIM + (size_t)g * cpg) * N_DIM;
    int t = threadIdx.x;

    float s = 0.f, ss = 0.f;
    for (int i = t; i < cpg * N_DIM; i += 256) { float v = xg[i]; s += v; ss += v * v; }

    __shared__ float rs[256], rss[256];
    rs[t] = s; rss[t] = ss; __syncthreads();
    for (int o = 128; o > 0; o >>= 1) {
        if (t < o) { rs[t] += rs[t + o]; rss[t] += rss[t + o]; }
        __syncthreads();
    }
    float inv_n = 1.0f / (float)(cpg * N_DIM);
    float mu  = rs[0] * inv_n;
    float var = rss[0] * inv_n - mu * mu;
    float inv = rsqrtf(var + 1e-6f);

    for (int i = t; i < cpg * N_DIM; i += 256) {
        int c = i / N_DIM, n = i % N_DIM;
        int cg = g * cpg + c;
        float hv = (xg[i] - mu) * inv * gamma[cg] + beta[cg];
        hT[((size_t)b * N_DIM + n) * C_DIM + cg] = (_Float16)hv;
    }
}

// ---------------------------------------------------------------------------
// 2) Weights -> f16 (q_w pre-scaled)
// ---------------------------------------------------------------------------
__global__ __launch_bounds__(256) void wconv_kernel(const float* __restrict__ qw,
                                                    const float* __restrict__ kw,
                                                    const float* __restrict__ vw,
                                                    const float* __restrict__ pw,
                                                    _Float16* __restrict__ wf) {
    int i = blockIdx.x * 256 + threadIdx.x;
    wf[i]                     = (_Float16)(qw[i] * QSCALE);
    wf[C_DIM * C_DIM + i]     = (_Float16)kw[i];
    wf[2 * C_DIM * C_DIM + i] = (_Float16)vw[i];
    wf[3 * C_DIM * C_DIM + i] = (_Float16)pw[i];
}

// ---------------------------------------------------------------------------
// 3) Fused Q/K/V conv.  Block = 4 waves, 64 n-rows.  A (hT) staged once via
//    TDM; weight panels (16x512) TDM double-buffered, consumed from LDS.
//    Each panel feeds 64 WMMAs.
// ---------------------------------------------------------------------------
__global__ __launch_bounds__(128) void qkv_kernel(const _Float16* __restrict__ hT,
                                                  const _Float16* __restrict__ wf,
                                                  const float* __restrict__ qb,
                                                  const float* __restrict__ kb,
                                                  const float* __restrict__ vb,
                                                  _Float16* __restrict__ qT,
                                                  _Float16* __restrict__ kT,
                                                  _Float16* __restrict__ v) {
    extern __shared__ _Float16 sm[];        // 64*LDSP (A) + 2*16*LDSP (panels)
    int b = blockIdx.x >> 6, n0 = (blockIdx.x & 63) * 64;
    const _Float16* hTb = hT + ((size_t)b * N_DIM + n0) * C_DIM;
    int t = threadIdx.x, wave = t >> 5, lane = t & 31;

#if HAVE_TDM
    if (wave == 0) {
        tdm_stage_rows(sm, hTb, 64, C_DIM);
        tdm_stage_rows(sm + 64 * LDSP, wf, 16, C_DIM);      // panel 0 (mat 0, o0 = 0)
    }
#else
    manual_stage_rows(sm, hTb, 64, C_DIM, t, 128);
#endif

    const _Float16* Amine = sm + (size_t)wave * 16 * LDSP;
    int col = lane & 15, rbase = (lane >> 4) * 8;
    int nw0 = n0 + wave * 16;

    for (int p = 0; p < 96; p++) {
        int mat = p >> 5, o0 = (p & 31) * 16;
        _Float16* Bcur = sm + (64 + (p & 1) * 16) * LDSP;
#if HAVE_TDM
        if (wave == 0) {
            if (p < 95) {
                int pn = p + 1;
                tdm_stage_rows(sm + (64 + (pn & 1) * 16) * LDSP,
                               wf + (size_t)(pn >> 5) * C_DIM * C_DIM + (size_t)(pn & 31) * 16 * C_DIM,
                               16, C_DIM);
                __builtin_amdgcn_s_wait_tensorcnt(1);
            } else {
                __builtin_amdgcn_s_wait_tensorcnt(0);
            }
        }
        __syncthreads();
#else
        __syncthreads();
        manual_stage_rows(Bcur, wf + (size_t)mat * C_DIM * C_DIM + (size_t)o0 * C_DIM,
                          16, C_DIM, t, 128);
        __syncthreads();
#endif
        float8 acc = {};
        #pragma unroll
        for (int k0 = 0; k0 < C_DIM; k0 += 32) {
            half16 a  = load_A_lds(Amine, k0, lane);
            half16 bm = load_B_lds(Bcur, k0, lane);
            acc = wmma_f16(a, bm, acc);
        }
        int o = o0 + col;
        if (mat == 0) {
            float bias = qb[o] * QSCALE;
            _Float16* dst = qT + (size_t)b * N_DIM * C_DIM;
            #pragma unroll
            for (int i = 0; i < 8; i++)
                dst[(size_t)(nw0 + rbase + i) * C_DIM + o] = (_Float16)(acc[i] + bias);
        } else if (mat == 1) {
            float bias = kb[o];
            _Float16* dst = kT + (size_t)b * N_DIM * C_DIM;
            #pragma unroll
            for (int i = 0; i < 8; i++)
                dst[(size_t)(nw0 + rbase + i) * C_DIM + o] = (_Float16)(acc[i] + bias);
        } else {
            float bias = vb[o];
            half8 hv;
            #pragma unroll
            for (int i = 0; i < 8; i++) hv[i] = (_Float16)(acc[i] + bias);
            *(half8*)(v + (size_t)b * C_DIM * N_DIM + (size_t)o * N_DIM + nw0 + rbase) = hv;
        }
        __syncthreads();    // all waves done with Bcur before re-fill
    }
}

// ---------------------------------------------------------------------------
// 4) S = qT . kT^T.  Block = 4 waves, 64 n-rows; K panels TDM double-buffered.
// ---------------------------------------------------------------------------
__global__ __launch_bounds__(128) void s_kernel(const _Float16* __restrict__ qT,
                                                const _Float16* __restrict__ kT,
                                                _Float16* __restrict__ S) {
    extern __shared__ _Float16 sm[];        // 64*LDSP (A) + 2*16*LDSP (panels)
    int b = blockIdx.x >> 6, n0 = (blockIdx.x & 63) * 64;
    const _Float16* qTb = qT + ((size_t)b * N_DIM + n0) * C_DIM;
    const _Float16* kTb = kT + (size_t)b * N_DIM * C_DIM;
    _Float16* Sb = S + (size_t)b * N_DIM * N_DIM;
    int t = threadIdx.x, wave = t >> 5, lane = t & 31;

#if HAVE_TDM
    if (wave == 0) {
        tdm_stage_rows(sm, qTb, 64, C_DIM);
        tdm_stage_rows(sm + 64 * LDSP, kTb, 16, C_DIM);
    }
#else
    manual_stage_rows(sm, qTb, 64, C_DIM, t, 128);
#endif

    const _Float16* Amine = sm + (size_t)wave * 16 * LDSP;
    int col = lane & 15, rbase = (lane >> 4) * 8;

    for (int mt = 0; mt < 256; mt++) {
        _Float16* Bcur = sm + (64 + (mt & 1) * 16) * LDSP;
#if HAVE_TDM
        if (wave == 0) {
            if (mt < 255) {
                tdm_stage_rows(sm + (64 + ((mt + 1) & 1) * 16) * LDSP,
                               kTb + (size_t)(mt + 1) * 16 * C_DIM, 16, C_DIM);
                __builtin_amdgcn_s_wait_tensorcnt(1);
            } else {
                __builtin_amdgcn_s_wait_tensorcnt(0);
            }
        }
        __syncthreads();
#else
        __syncthreads();
        manual_stage_rows(Bcur, kTb + (size_t)mt * 16 * C_DIM, 16, C_DIM, t, 128);
        __syncthreads();
#endif
        float8 acc = {};
        #pragma unroll
        for (int k0 = 0; k0 < C_DIM; k0 += 32) {
            half16 a  = load_A_lds(Amine, k0, lane);
            half16 bm = load_B_lds(Bcur, k0, lane);
            acc = wmma_f16(a, bm, acc);
        }
        int m0 = mt * 16;
        #pragma unroll
        for (int i = 0; i < 8; i++)
            Sb[(size_t)(n0 + wave * 16 + rbase + i) * N_DIM + m0 + col] = (_Float16)acc[i];
        __syncthreads();
    }
}

// ---------------------------------------------------------------------------
// 5) Row softmax in place over m.
// ---------------------------------------------------------------------------
__global__ __launch_bounds__(256) void softmax_kernel(_Float16* __restrict__ S) {
    _Float16* p = S + (size_t)blockIdx.x * N_DIM;
    int t = threadIdx.x;
    half16 hv = *(const half16*)(p + t * 16);
    float v[16];
    float mx = -3.0e38f;
    #pragma unroll
    for (int i = 0; i < 16; i++) { v[i] = (float)hv[i]; mx = fmaxf(mx, v[i]); }

    __shared__ float red[256];
    red[t] = mx; __syncthreads();
    for (int o = 128; o > 0; o >>= 1) {
        if (t < o) red[t] = fmaxf(red[t], red[t + o]);
        __syncthreads();
    }
    mx = red[0]; __syncthreads();

    float s = 0.f;
    #pragma unroll
    for (int i = 0; i < 16; i++) { v[i] = __expf(v[i] - mx); s += v[i]; }
    red[t] = s; __syncthreads();
    for (int o = 128; o > 0; o >>= 1) {
        if (t < o) red[t] += red[t + o];
        __syncthreads();
    }
    float inv = 1.0f / red[0];

    half16 ov;
    #pragma unroll
    for (int i = 0; i < 16; i++) ov[i] = (_Float16)(v[i] * inv);
    *(half16*)(p + t * 16) = ov;
}

// ---------------------------------------------------------------------------
// 6) h_attnT[b,n,c] = sum_m P[n,m] v[c,m].  Block = 16 waves, 64 n-rows.
//    P chunks (64x512) TDM double-buffered.  Wave = (n-subtile, c-quarter):
//    8 accumulator tiles -> 8 independent B-load chains per k-step.
// ---------------------------------------------------------------------------
__global__ __launch_bounds__(512) void pv_kernel(const _Float16* __restrict__ S,
                                                 const _Float16* __restrict__ v,
                                                 _Float16* __restrict__ hattnT) {
    extern __shared__ _Float16 sm[];        // 2 * 64*LDSP
    int b = blockIdx.x >> 6, n0 = (blockIdx.x & 63) * 64;
    const _Float16* Sb = S + ((size_t)b * N_DIM + n0) * N_DIM;
    const _Float16* vb = v + (size_t)b * C_DIM * N_DIM;
    int t = threadIdx.x, wave = t >> 5, lane = t & 31;
    int nsub = wave & 3, cq = wave >> 2;    // 4 n-subtiles x 4 c-quarters

#if HAVE_TDM
    if (wave == 0) tdm_stage_rows(sm, Sb, 64, N_DIM);
#endif

    float8 acc[8];
    #pragma unroll
    for (int j = 0; j < 8; j++) acc[j] = (float8){};

    for (int ch = 0; ch < 8; ch++) {
        int mc = ch * 512;
        _Float16* Ac = sm + (ch & 1) * 64 * LDSP;
#if HAVE_TDM
        if (wave == 0) {
            if (ch < 7) {
                tdm_stage_rows(sm + ((ch + 1) & 1) * 64 * LDSP,
                               Sb + (size_t)(ch + 1) * 512, 64, N_DIM);
                __builtin_amdgcn_s_wait_tensorcnt(1);
            } else {
                __builtin_amdgcn_s_wait_tensorcnt(0);
            }
        }
        __syncthreads();
#else
        __syncthreads();
        manual_stage_rows(Ac, Sb + (size_t)mc, 64, N_DIM, t, 512);
        __syncthreads();
#endif
        const _Float16* Aw = Ac + (size_t)nsub * 16 * LDSP;
        #pragma unroll 4
        for (int k0 = 0; k0 < 512; k0 += 32) {
            half16 a = load_A_lds(Aw, k0, lane);
            half16 bm[8];
            #pragma unroll
            for (int j = 0; j < 8; j++) {
                int c0 = (cq * 8 + j) * 16;
                bm[j] = load_B_glb(vb + (size_t)c0 * N_DIM + mc + k0, N_DIM, lane);
            }
            #pragma unroll
            for (int j = 0; j < 8; j++) acc[j] = wmma_f16(a, bm[j], acc[j]);
        }
        __syncthreads();    // done with Ac before re-fill
    }

    int col = lane & 15, rbase = (lane >> 4) * 8;
    #pragma unroll
    for (int j = 0; j < 8; j++) {
        int c0 = (cq * 8 + j) * 16;
        #pragma unroll
        for (int i = 0; i < 8; i++)
            hattnT[((size_t)b * N_DIM + n0 + nsub * 16 + rbase + i) * C_DIM + c0 + col]
                = (_Float16)acc[j][i];
    }
}

// ---------------------------------------------------------------------------
// 7) proj conv + bias + residual.  Same panel scheme as qkv, 32 panels.
// ---------------------------------------------------------------------------
__global__ __launch_bounds__(128) void proj_kernel(const _Float16* __restrict__ hattnT,
                                                   const _Float16* __restrict__ pwf,
                                                   const float* __restrict__ pb,
                                                   const float* __restrict__ x,
                                                   float* __restrict__ out) {
    extern __shared__ _Float16 sm[];        // 64*LDSP (A) + 2*16*LDSP (panels)
    int b = blockIdx.x >> 6, n0 = (blockIdx.x & 63) * 64;
    const _Float16* hb = hattnT + ((size_t)b * N_DIM + n0) * C_DIM;
    int t = threadIdx.x, wave = t >> 5, lane = t & 31;

#if HAVE_TDM
    if (wave == 0) {
        tdm_stage_rows(sm, hb, 64, C_DIM);
        tdm_stage_rows(sm + 64 * LDSP, pwf, 16, C_DIM);
    }
#else
    manual_stage_rows(sm, hb, 64, C_DIM, t, 128);
#endif

    const _Float16* Amine = sm + (size_t)wave * 16 * LDSP;
    int col = lane & 15, rbase = (lane >> 4) * 8;
    int nw0 = n0 + wave * 16;

    for (int ot = 0; ot < 32; ot++) {
        int o0 = ot * 16;
        _Float16* Bcur = sm + (64 + (ot & 1) * 16) * LDSP;
#if HAVE_TDM
        if (wave == 0) {
            if (ot < 31) {
                tdm_stage_rows(sm + (64 + ((ot + 1) & 1) * 16) * LDSP,
                               pwf + (size_t)(ot + 1) * 16 * C_DIM, 16, C_DIM);
                __builtin_amdgcn_s_wait_tensorcnt(1);
            } else {
                __builtin_amdgcn_s_wait_tensorcnt(0);
            }
        }
        __syncthreads();
#else
        __syncthreads();
        manual_stage_rows(Bcur, pwf + (size_t)o0 * C_DIM, 16, C_DIM, t, 128);
        __syncthreads();
#endif
        float8 acc = {};
        #pragma unroll
        for (int k0 = 0; k0 < C_DIM; k0 += 32) {
            half16 a  = load_A_lds(Amine, k0, lane);
            half16 bm = load_B_lds(Bcur, k0, lane);
            acc = wmma_f16(a, bm, acc);
        }
        int o = o0 + col;
        float bias = pb[o];
        size_t base = ((size_t)b * C_DIM + o) * N_DIM + nw0 + rbase;
        float8 xr = *(const float8*)(x + base);
        float8 r;
        #pragma unroll
        for (int i = 0; i < 8; i++) r[i] = xr[i] + acc[i] + bias;
        *(float8*)(out + base) = r;
        __syncthreads();
    }
}

// ---------------------------------------------------------------------------
// Host launcher
// ---------------------------------------------------------------------------
extern "C" void kernel_launch(void* const* d_in, const int* in_sizes, int n_in,
                              void* d_out, int out_size, void* d_ws, size_t ws_size,
                              hipStream_t stream) {
    const float* x     = (const float*)d_in[0];
    const float* gamma = (const float*)d_in[1];
    const float* beta  = (const float*)d_in[2];
    const float* q_w   = (const float*)d_in[3];
    const float* q_b   = (const float*)d_in[4];
    const float* k_w   = (const float*)d_in[5];
    const float* k_b   = (const float*)d_in[6];
    const float* v_w   = (const float*)d_in[7];
    const float* v_b   = (const float*)d_in[8];
    const float* p_w   = (const float*)d_in[9];
    const float* p_b   = (const float*)d_in[10];
    float* out = (float*)d_out;

    char* ws = (char*)d_ws;
    size_t off = 0;
    auto take = [&](size_t bytes) { char* p = ws + off; off = (off + bytes + 255) & ~(size_t)255; return p; };
    _Float16* wf     = (_Float16*)take((size_t)4 * C_DIM * C_DIM * 2);      //   2 MB
    _Float16* hT     = (_Float16*)take((size_t)B_DIM * N_DIM * C_DIM * 2);  //  32 MB
    _Float16* qT     = (_Float16*)take((size_t)B_DIM * N_DIM * C_DIM * 2);  //  32 MB
    _Float16* kT     = (_Float16*)take((size_t)B_DIM * N_DIM * C_DIM * 2);  //  32 MB
    _Float16* vmat   = (_Float16*)take((size_t)B_DIM * C_DIM * N_DIM * 2);  //  32 MB
    _Float16* hattnT = (_Float16*)take((size_t)B_DIM * N_DIM * C_DIM * 2);  //  32 MB
    _Float16* S      = (_Float16*)take((size_t)B_DIM * N_DIM * N_DIM * 2);  // 256 MB

    const size_t lds_panels = (size_t)(64 + 32) * LDSP * 2;   // A + 2 x 16-row panels
    const size_t lds_pv     = (size_t)2 * 64 * LDSP * 2;      // 2 x 64-row P chunks

    gn_kernel     <<<B_DIM * 32,           256, 0,          stream>>>(x, gamma, beta, hT);
    wconv_kernel  <<<C_DIM * C_DIM / 256,  256, 0,          stream>>>(q_w, k_w, v_w, p_w, wf);
    qkv_kernel    <<<B_DIM * (N_DIM / 64), 128, lds_panels, stream>>>(hT, wf, q_b, k_b, v_b, qT, kT, vmat);
    s_kernel      <<<B_DIM * (N_DIM / 64), 128, lds_panels, stream>>>(qT, kT, S);
    softmax_kernel<<<B_DIM * N_DIM,        256, 0,          stream>>>(S);
    pv_kernel     <<<B_DIM * (N_DIM / 64), 512, lds_pv,     stream>>>(S, vmat, hattnT);
    proj_kernel   <<<B_DIM * (N_DIM / 64), 128, lds_panels, stream>>>(hattnT, wf + (size_t)3 * C_DIM * C_DIM, p_b, x, out);
}
